// get_edge_feature_13237089206320
// MI455X (gfx1250) — compile-verified, matrix operand order checked by reading, not run on hardware
//
#include <hip/hip_runtime.h>

typedef __attribute__((ext_vector_type(2)))  float    v2f;
typedef __attribute__((ext_vector_type(8)))  float    v8f;
typedef __attribute__((ext_vector_type(16))) _Float16 v16h;
typedef int v4i_ __attribute__((vector_size(16)));   // matches builtin param type
typedef __attribute__((address_space(1))) v4i_* gp128_t;
typedef __attribute__((address_space(3))) v4i_* lp128_t;

constexpr int Bc = 8;
constexpr int Nc = 4096;
constexpr int Kc = 16;
constexpr int EF = Bc * 6 * Kc * 3 * Nc;   // 9,437,184 edge-feature elements

#define BIGF 3.0e38f

// ---------------------------------------------------------------------------
// Phase 1: fused pairwise-distance (WMMA f32 16x16x4) + top-16 selection.
// Block = 256 threads (8 waves, wave32). Each wave owns 16 query points.
// grid = (N/128, B).
// ---------------------------------------------------------------------------
__global__ __launch_bounds__(256)
void knn_wmma_kernel(const float* __restrict__ pc, int* __restrict__ idxOut)
{
    // refs: (3, 4096) staged points (48KB) | dot: 8 waves * 16x16 f32 tiles (8KB)
    __shared__ float smem[3 * Nc + 8 * 256];
    float* refs = smem;
    float* dotb = smem + 3 * Nc;

    const int  tid  = threadIdx.x;
    const int  b    = blockIdx.y;
    const int  wave = tid >> 5;
    const int  lane = tid & 31;
    const int  ln   = lane & 15;
    const bool lo   = (lane < 16);

    // Stage this batch's point cloud into LDS.
    const float* pcb = pc + (size_t)b * 3 * Nc;
#if __has_builtin(__builtin_amdgcn_global_load_async_to_lds_b128)
    {
        // CDNA5 async bulk copy: global -> LDS with no VGPR round trip
        // (tracked by ASYNCcnt). 3072 x b128 transfers, 12 per thread.
        __attribute__((address_space(1))) char* g =
            (__attribute__((address_space(1))) char*)pcb;           // const cast away
        __attribute__((address_space(3))) char* l =
            (__attribute__((address_space(3))) char*)refs;
        for (int t = tid; t < (3 * Nc) / 4; t += 256) {
            __builtin_amdgcn_global_load_async_to_lds_b128(
                (gp128_t)(g + 16 * t), (lp128_t)(l + 16 * t), 0, 0);
        }
        asm volatile("s_wait_asynccnt 0" ::: "memory");
    }
#else
    for (int t = tid; t < 3 * Nc; t += 256) refs[t] = pcb[t];
#endif
    __syncthreads();

    const int qbase = blockIdx.x * 128 + wave * 16;
    const int q     = qbase + ln;

    // A operand: 16 queries x K=4 (x,y,z,0), pre-scaled by -2 so the WMMA
    // directly produces  -2*dot(q,r) + |r|^2  (the per-row |q|^2 term is a
    // row-constant and cannot change per-row top-k ranking).
#if __has_builtin(__builtin_amdgcn_wmma_f32_16x16x4_f32)
    // f32 A 16x4 layout: VGPR0 = K0 (lanes 0-15) / K2 (lanes 16-31),
    //                    VGPR1 = K1 (lanes 0-15) / K3 (lanes 16-31).
    v2f Aop;
    Aop.x = -2.0f * (lo ? refs[q] : refs[2 * Nc + q]);
    Aop.y = lo ? (-2.0f * refs[Nc + q]) : 0.0f;
#else
    v16h Aop = {};
    if (lo) {
        Aop[0] = (_Float16)(-2.0f * refs[q]);
        Aop[1] = (_Float16)(-2.0f * refs[Nc + q]);
        Aop[2] = (_Float16)(-2.0f * refs[2 * Nc + q]);
    }
#endif

    // Per-lane ascending top-16; two lanes per query row, 8 columns each.
    float best[16];
    int   bidx[16];
#pragma unroll
    for (int t = 0; t < 16; ++t) { best[t] = BIGF; bidx[t] = 0; }

    float*    dt      = dotb + wave * 256;
    const int rbase   = lo ? 0 : 8;
    const int colBase = lo ? 0 : 8;
    const int row     = ln;
    const int myq     = qbase + row;

    for (int j = 0; j < Nc / 16; ++j) {
        const int   rn = j * 16 + ln;
        const float rx = refs[rn];
        const float ry = refs[Nc + rn];
        const float rz = refs[2 * Nc + rn];
        const float sq = __builtin_fmaf(rx, rx, __builtin_fmaf(ry, ry, rz * rz));

        // C operand: every C/D VGPR of column n holds |r_n|^2.
        v8f cacc;
#pragma unroll
        for (int r = 0; r < 8; ++r) cacc[r] = sq;

#if __has_builtin(__builtin_amdgcn_wmma_f32_16x16x4_f32)
        v2f Bop;
        Bop.x = lo ? rx : rz;       // K0 / K2 rows
        Bop.y = lo ? ry : 0.0f;     // K1 / K3(pad) rows
        v8f D = __builtin_amdgcn_wmma_f32_16x16x4_f32(
            false, Aop, false, Bop, (short)0, cacc, false, false);
#else
        v16h Bop = {};
        if (lo) { Bop[0] = (_Float16)rx; Bop[1] = (_Float16)ry; Bop[2] = (_Float16)rz; }
        v8f D = __builtin_amdgcn_wmma_f32_16x16x32_f16(
            false, Aop, false, Bop, (short)0, cacc, false, false);
#endif

        // Scatter 16x16 tile to LDS: VGPR r -> row (rbase+r), col ln.
#pragma unroll
        for (int r = 0; r < 8; ++r) dt[(rbase + r) * 16 + ln] = D[r];
        asm volatile("s_wait_dscnt 0" ::: "memory");   // wave-local LDS RAW fence

        // Selection: this lane scans 8 candidates of its query row.
        const float* drow = dt + row * 16;
#pragma unroll 4
        for (int c = 0; c < 8; ++c) {
            const int col  = colBase + c;
            const int ridx = j * 16 + col;
            float v  = drow[col];
            v = (ridx == myq) ? BIGF : v;          // exclude self (select, no branch)
            if (v < best[15]) {
                // Branchless bubble insert: one cmp + min/max/cndmask per slot.
                int vi = ridx;
#pragma unroll
                for (int t = 0; t < 16; ++t) {
                    const bool  lt   = v < best[t];
                    const float nmin = fminf(v, best[t]);
                    const float nmax = fmaxf(v, best[t]);
                    const int   imin = lt ? vi : bidx[t];
                    const int   imax = lt ? bidx[t] : vi;
                    best[t] = nmin; bidx[t] = imin;
                    v = nmax; vi = imax;
                }
            }
        }
        asm volatile("" ::: "memory");
    }

    // Merge the two per-lane sorted lists of each query row; write idx (B,K,N).
    __syncthreads();                         // refs region is reused below
    float* mD = smem;                        // 8*32*16 floats
    int*   mI = (int*)(smem + 4096);         // 8*32*16 ints
    {
        float* md = mD + (wave * 32 + lane) * 16;
        int*   mi = mI + (wave * 32 + lane) * 16;
#pragma unroll
        for (int t = 0; t < 16; ++t) { md[t] = best[t]; mi[t] = bidx[t]; }
    }
    asm volatile("s_wait_dscnt 0" ::: "memory");

    if (lo) {
        const float* aD = mD + (wave * 32 + ln) * 16;
        const float* bD = mD + (wave * 32 + ln + 16) * 16;
        const int*   aI = mI + (wave * 32 + ln) * 16;
        const int*   bI = mI + (wave * 32 + ln + 16) * 16;
        int ia = 0, ib = 0;
        const int n = qbase + ln;
        for (int k = 0; k < 16; ++k) {
            const float av = aD[ia];
            const float bv = bD[ib];
            const bool  ta = (av <= bv);
            const int   id = ta ? aI[ia] : bI[ib];
            ia += ta ? 1 : 0;
            ib += ta ? 0 : 1;
            idxOut[(b * Kc + k) * Nc + n] = id;
        }
    }
}

// ---------------------------------------------------------------------------
// Phase 2: edge feature gather. out shape (B, 6, K, 3, N).
//   c<3 : central = input[b,c,r,n]
//   c>=3: pc[b,c-3,idx[b,k,n]] - input[b,c-3,r,n]
// ---------------------------------------------------------------------------
__global__ __launch_bounds__(256)
void edge_feature_kernel(const float* __restrict__ pc, const float* __restrict__ inp,
                         const int* __restrict__ idxI, float* __restrict__ out)
{
    int i = blockIdx.x * 256 + threadIdx.x;
    if (i >= EF) return;
    const int n = i & (Nc - 1);
    int t = i >> 12;                 // / 4096
    const int r = t % 3;  t /= 3;
    const int k = t & (Kc - 1); t >>= 4;
    const int c = t % 6;
    const int b = t / 6;
    const int cc = (c < 3) ? c : (c - 3);

    const float central = inp[(((b * 3 + cc) * 3 + r) << 12) + n];
    float v;
    if (c < 3) {
        v = central;
    } else {
        const int id = idxI[((b * Kc + k) << 12) + n];
        v = pc[((b * 3 + cc) << 12) + id] - central;
    }
    out[i] = v;
}

// ---------------------------------------------------------------------------
extern "C" void kernel_launch(void* const* d_in, const int* in_sizes, int n_in,
                              void* d_out, int out_size, void* d_ws, size_t ws_size,
                              hipStream_t stream)
{
    (void)in_sizes; (void)n_in; (void)out_size; (void)d_ws; (void)ws_size;

    const float* pc  = (const float*)d_in[0];   // (8, 3, 4096) f32
    const float* inp = (const float*)d_in[1];   // (8, 3, 3, 4096) f32
    float* out    = (float*)d_out;              // edge_feature (B,6,K,3,N)
    int*   idxOut = (int*)d_out + EF;           // idx (B,K,N) int32, concatenated

    dim3 g1(Nc / 128, Bc);
    knn_wmma_kernel<<<g1, 256, 0, stream>>>(pc, idxOut);

    edge_feature_kernel<<<(EF + 255) / 256, 256, 0, stream>>>(pc, inp, idxOut, out);
}
